// DeepSeekSparseMLHA_69449621176865
// MI455X (gfx1250) — matrix operand order repaired and checked by reading, not compiled, not run on hardware
//
#include <hip/hip_runtime.h>

// ---------------- model dims ----------------
#define T_SEQ   2048
#define D_MODEL 2048
#define HI_N    4
#define DI_N    64
#define H_N     16
#define DH_N    128
#define TOPK    512
#define NEGF    (-1e9f)

typedef __attribute__((ext_vector_type(16))) __bf16 v16bf;
typedef __attribute__((ext_vector_type(8)))  __bf16 v8bf;
typedef __attribute__((ext_vector_type(8)))  float  v8f;

// ---------------- WMMA fragment helpers (CDNA5 16x16x32 bf16 layout) ----------------
__device__ inline v16bf combine16(v8bf lo, v8bf hi) {
  v16bf r;
#pragma unroll
  for (int i = 0; i < 8; i++) { r[i] = lo[i]; r[i + 8] = hi[i]; }
  return r;
}

// A-matrix 16x32: lane holds row M=lane&15; khalf=lane>>4 selects K={kh*8..} U {16+kh*8..}
__device__ inline v16bf load_a_frag(const __bf16* rowptr_k0, int lhalf) {
  v8bf a0 = *(const v8bf*)(rowptr_k0 + lhalf * 8);
  v8bf a1 = *(const v8bf*)(rowptr_k0 + 16 + lhalf * 8);
  return combine16(a0, a1);
}

// B-matrix 32x16 from pre-transposed BT[N][K]: lane holds col N=lane&15; khalf=lane>>4 selects K={kh*8..} U {16+kh*8..}
__device__ inline v16bf load_b_frag(const __bf16* rowptr_k0, int lhalf) {
  const __bf16* q = rowptr_k0 + lhalf * 8;
  v8bf b0 = *(const v8bf*)q;
  v8bf b1 = *(const v8bf*)(q + 16);
  return combine16(b0, b1);
}

__device__ inline float wave_max32(float v) {
#pragma unroll
  for (int m = 16; m > 0; m >>= 1) v = fmaxf(v, __shfl_xor(v, m, 32));
  return v;
}
__device__ inline float wave_sum32(float v) {
#pragma unroll
  for (int m = 16; m > 0; m >>= 1) v += __shfl_xor(v, m, 32);
  return v;
}

// ---------------- elementwise f32 -> bf16 ----------------
__global__ __launch_bounds__(256) void cvt_bf16_kernel(const float* __restrict__ src,
                                                       __bf16* __restrict__ dst, int n) {
  int i = blockIdx.x * 256 + threadIdx.x;
  if (i < n) dst[i] = (__bf16)src[i];
}

// ---------------- W[K][N] f32 -> WT[Npad][K] bf16 (zero-padded) ----------------
__global__ __launch_bounds__(256) void transpose_w_kernel(const float* __restrict__ W,
                                                          __bf16* __restrict__ WT,
                                                          int K, int N, int Npad) {
  int idx = blockIdx.x * 256 + threadIdx.x;
  int total = Npad * K;
  if (idx >= total) return;
  int n = idx / K;
  int k = idx - n * K;
  WT[(size_t)n * K + k] = (n < N) ? (__bf16)W[(size_t)k * N + n] : (__bf16)0.0f;
}

// ---------------- tiny projection: wts[t][h] = x[t,:] . Ww[:,h] ----------------
__global__ __launch_bounds__(256) void w_proj_kernel(const float* __restrict__ x,
                                                     const float* __restrict__ Ww,
                                                     float* __restrict__ wts) {
  int id = blockIdx.x * 256 + threadIdx.x;
  if (id >= T_SEQ * HI_N) return;
  int t = id >> 2, h = id & 3;
  float acc = 0.f;
  for (int k = 0; k < D_MODEL; k++) acc += x[(size_t)t * D_MODEL + k] * Ww[k * HI_N + h];
  wts[id] = acc;
}

// ---------------- bf16 WMMA GEMM: C[M,N] = A[M,K] * BT[N,K]^T ----------------
// 256 threads = 8 waves laid out 4(M) x 2(N); block tile 128(M) x 128(N).
// Wave tile 32(M) x 64(N) = 2x4 accumulators; ping-pong K pipeline (no register copies).
// Requires K % 64 == 0.
__global__ __launch_bounds__(256) void gemm_bf16_wmma(const __bf16* __restrict__ A,
                                                      const __bf16* __restrict__ BT,
                                                      void* __restrict__ Cv,
                                                      int M, int N, int K, int ldc,
                                                      int outBf16) {
  const int lane = threadIdx.x & 31;
  const int wv   = threadIdx.x >> 5;
  const int mw = wv & 3, nw = wv >> 2;
  const int m0 = blockIdx.y * 128 + mw * 32;
  const int n0 = blockIdx.x * 128 + nw * 64;
  const int lrow = lane & 15, lhalf = lane >> 4;

  v8f acc[2][4];
#pragma unroll
  for (int mi = 0; mi < 2; mi++)
#pragma unroll
    for (int j = 0; j < 4; j++) acc[mi][j] = (v8f)0.0f;

  const __bf16* ap0 = A + (size_t)(m0 + lrow) * K;
  const __bf16* ap1 = ap0 + (size_t)16 * K;
  const __bf16* bp[4];
#pragma unroll
  for (int j = 0; j < 4; j++) bp[j] = BT + (size_t)(n0 + j * 16 + lrow) * K;

  auto mma8 = [&](const v16bf& x0, const v16bf& x1, const v16bf (&bb)[4]) {
#pragma unroll
    for (int j = 0; j < 4; j++) {
      acc[0][j] = __builtin_amdgcn_wmma_f32_16x16x32_bf16(false, x0, false, bb[j],
                                                          (short)0, acc[0][j], false, false);
      acc[1][j] = __builtin_amdgcn_wmma_f32_16x16x32_bf16(false, x1, false, bb[j],
                                                          (short)0, acc[1][j], false, false);
    }
  };

  // ping-pong buffers
  v16bf a0A = load_a_frag(ap0, lhalf);
  v16bf a1A = load_a_frag(ap1, lhalf);
  v16bf bA[4];
#pragma unroll
  for (int j = 0; j < 4; j++) bA[j] = load_b_frag(bp[j], lhalf);
  v16bf a0B, a1B, bB[4];

  int k0 = 0;
  for (; k0 < K - 64; k0 += 64) {
    const int k1 = k0 + 32;
    a0B = load_a_frag(ap0 + k1, lhalf);
    a1B = load_a_frag(ap1 + k1, lhalf);
#pragma unroll
    for (int j = 0; j < 4; j++) bB[j] = load_b_frag(bp[j] + k1, lhalf);
    __builtin_prefetch(ap0 + k0 + 128, 0, 3);
    mma8(a0A, a1A, bA);   // overlaps the k1 loads above

    const int k2 = k0 + 64;
    a0A = load_a_frag(ap0 + k2, lhalf);
    a1A = load_a_frag(ap1 + k2, lhalf);
#pragma unroll
    for (int j = 0; j < 4; j++) bA[j] = load_b_frag(bp[j] + k2, lhalf);
    mma8(a0B, a1B, bB);   // overlaps the k2 loads above
  }
  // tail: last 64 of K (k0 == K-64)
  {
    const int k1 = k0 + 32;
    a0B = load_a_frag(ap0 + k1, lhalf);
    a1B = load_a_frag(ap1 + k1, lhalf);
#pragma unroll
    for (int j = 0; j < 4; j++) bB[j] = load_b_frag(bp[j] + k1, lhalf);
    mma8(a0A, a1A, bA);
    mma8(a0B, a1B, bB);
  }

#pragma unroll
  for (int mi = 0; mi < 2; mi++) {
#pragma unroll
    for (int j = 0; j < 4; j++) {
      int n = n0 + j * 16 + lrow;
      if (n >= N) continue;
#pragma unroll
      for (int r = 0; r < 8; r++) {
        int m = m0 + mi * 16 + r + lhalf * 8;
        float v = acc[mi][j][r];
        if (outBf16) ((__bf16*)Cv)[(size_t)m * ldc + n] = (__bf16)v;
        else         ((float*)Cv)[(size_t)m * ldc + n] = v;
      }
    }
  }
}

// ---------------- indexer scores (WMMA over DI): masked scores row per query ----------------
// block = 16 queries, 512 threads = 16 waves; wave strides over 16-wide key tiles.
// A fragments (qi) and w[t,h] are s-invariant: hoisted. B (ki) is shared by all 4 heads.
__global__ __launch_bounds__(512) void index_score_kernel(const __bf16* __restrict__ qi,
                                                          const __bf16* __restrict__ ki,
                                                          const float* __restrict__ wts,
                                                          float* __restrict__ sco) {
  const int lane = threadIdx.x & 31;
  const int wv   = threadIdx.x >> 5;
  const int lrow = lane & 15, lhalf = lane >> 4;
  const int bx = blockIdx.x;
  const int t0 = bx * 16;

  int trow[8];
#pragma unroll
  for (int r = 0; r < 8; r++) trow[r] = t0 + r + lhalf * 8;

  // hoisted A fragments: [head][kchunk]
  v16bf av[HI_N][2];
#pragma unroll
  for (int h = 0; h < HI_N; h++)
#pragma unroll
    for (int kc = 0; kc < 2; kc++)
      av[h][kc] = load_a_frag(qi + (size_t)(t0 + lrow) * (HI_N * DI_N) + h * DI_N + kc * 32,
                              lhalf);
  // hoisted per-row head weights
  float wq[HI_N][8];
#pragma unroll
  for (int h = 0; h < HI_N; h++)
#pragma unroll
    for (int r = 0; r < 8; r++) wq[h][r] = wts[trow[r] * HI_N + h];

  for (int st = wv; st < T_SEQ / 16; st += 16) {
    const int s0 = st * 16;
    const int scol = s0 + lrow;
    if (st > bx) {  // fully above diagonal -> NEG
#pragma unroll
      for (int r = 0; r < 8; r++) sco[(size_t)trow[r] * T_SEQ + scol] = NEGF;
      continue;
    }
    // B fragments for this key tile (shared across heads): 4 x b128 for 8 WMMAs
    const __bf16* bp = ki + (size_t)(s0 + lrow) * DI_N;
    v16bf bv0 = load_b_frag(bp, lhalf);
    v16bf bv1 = load_b_frag(bp + 32, lhalf);

    float tot[8] = {0.f, 0.f, 0.f, 0.f, 0.f, 0.f, 0.f, 0.f};
#pragma unroll
    for (int h = 0; h < HI_N; h++) {
      v8f d = (v8f)0.0f;
      d = __builtin_amdgcn_wmma_f32_16x16x32_bf16(false, av[h][0], false, bv0,
                                                  (short)0, d, false, false);
      d = __builtin_amdgcn_wmma_f32_16x16x32_bf16(false, av[h][1], false, bv1,
                                                  (short)0, d, false, false);
#pragma unroll
      for (int r = 0; r < 8; r++) {
        float rv = d[r] > 0.f ? d[r] : 0.f;   // relu
        tot[r] += rv * wq[h][r];              // * w[t,h]
      }
    }
#pragma unroll
    for (int r = 0; r < 8; r++) {
      float v = (scol <= trow[r]) ? tot[r] : NEGF;  // causal mask
      sco[(size_t)trow[r] * T_SEQ + scol] = v;
    }
  }
}

// ---------------- top-k per query via threshold bisection in LDS ----------------
__global__ __launch_bounds__(256) void topk_kernel(const float* __restrict__ sco,
                                                   int* __restrict__ top_idx) {
  const int t = blockIdx.x;
  const int tid = threadIdx.x;
  if (t < TOPK) {  // short rows: selected set is exactly {0..511} (reference tie-break quirk)
    for (int i = tid; i < TOPK; i += 256) top_idx[(size_t)t * TOPK + i] = i;
    return;
  }
  __shared__ float row[T_SEQ];
  __shared__ float red[256];
  __shared__ int cnt;
  __shared__ int wpos;
  __shared__ float slo, shi;

  for (int i = tid; i < T_SEQ; i += 256) row[i] = sco[(size_t)t * T_SEQ + i];
  __syncthreads();

  float mx = -3.4e38f;
  for (int i = tid; i < T_SEQ; i += 256) mx = fmaxf(mx, row[i]);
  red[tid] = mx; __syncthreads();
  if (tid == 0) { float m = red[0]; for (int i = 1; i < 256; i++) m = fmaxf(m, red[i]); shi = m; }
  __syncthreads();
  float mn = 3.4e38f;
  for (int i = tid; i < T_SEQ; i += 256) { float v = row[i]; if (v > -1e8f) mn = fminf(mn, v); }
  red[tid] = mn; __syncthreads();
  if (tid == 0) { float m = red[0]; for (int i = 1; i < 256; i++) m = fminf(m, red[i]); slo = m - 1.0f; }
  __syncthreads();

  float lo = slo, hi = shi;   // invariant: count(>lo) > K >= count(>hi)
  for (int it = 0; it < 40; it++) {
    float mid = 0.5f * (lo + hi);
    if (tid == 0) cnt = 0;
    __syncthreads();
    int c = 0;
    for (int i = tid; i < T_SEQ; i += 256) c += (row[i] > mid) ? 1 : 0;
    atomicAdd(&cnt, c);
    __syncthreads();
    int total = cnt;
    if (total > TOPK) lo = mid; else hi = mid;
    __syncthreads();
  }

  if (tid == 0) wpos = 0;
  __syncthreads();
  for (int i = tid; i < T_SEQ; i += 256) {
    if (row[i] > hi) {
      int p = atomicAdd(&wpos, 1);
      if (p < TOPK) top_idx[(size_t)t * TOPK + p] = i;
    }
  }
  __syncthreads();
  for (int i = tid; i < T_SEQ; i += 256) {
    float v = row[i];
    if (v > lo && v <= hi) {
      int p = atomicAdd(&wpos, 1);
      if (p < TOPK) top_idx[(size_t)t * TOPK + p] = i;
    }
  }
  __syncthreads();
  if (tid == 0) { int p = wpos; while (p < TOPK) top_idx[(size_t)t * TOPK + p++] = t; }
}

// ---------------- per-query sparse attention (gather-bound -> VALU) ----------------
__global__ __launch_bounds__(256) void attn_kernel(const __bf16* __restrict__ qb,
                                                   const __bf16* __restrict__ kb,
                                                   const __bf16* __restrict__ vb,
                                                   const int* __restrict__ top_idx,
                                                   __bf16* __restrict__ attnb) {
  const int t = blockIdx.x;
  const int tid = threadIdx.x;
  __shared__ int sel[TOPK];
  __shared__ float qrow[D_MODEL];
  __shared__ float sc[H_N * TOPK];   // 32 KB

  for (int i = tid; i < TOPK; i += 256) sel[i] = top_idx[(size_t)t * TOPK + i];
  for (int i = tid; i < D_MODEL; i += 256) qrow[i] = (float)qb[(size_t)t * D_MODEL + i];
  __syncthreads();

  const float scale = 0.08838834764831845f;  // 1/sqrt(128)
  for (int pr = tid; pr < H_N * TOPK; pr += 256) {
    int h = pr >> 9;
    int j = pr & (TOPK - 1);
    int s = sel[j];
    const __bf16* kp = kb + (size_t)s * D_MODEL + h * DH_N;
    float acc = 0.f;
#pragma unroll
    for (int c = 0; c < DH_N / 8; c++) {
      v8bf kk = *(const v8bf*)(kp + c * 8);
#pragma unroll
      for (int e = 0; e < 8; e++) acc += (float)kk[e] * qrow[h * DH_N + c * 8 + e];
    }
    sc[pr] = acc * scale;
  }
  __syncthreads();

  // per-head softmax: 8 waves, wave w owns rows w and w+8
  const int lane = threadIdx.x & 31;
  const int wv   = threadIdx.x >> 5;
  for (int h = wv; h < H_N; h += 8) {
    float mx = -3.4e38f;
    for (int j = lane; j < TOPK; j += 32) mx = fmaxf(mx, sc[h * TOPK + j]);
    mx = wave_max32(mx);
    float sum = 0.f;
    for (int j = lane; j < TOPK; j += 32) {
      float e = __expf(sc[h * TOPK + j] - mx);
      sc[h * TOPK + j] = e;
      sum += e;
    }
    sum = wave_sum32(sum);
    float inv = 1.f / sum;
    for (int j = lane; j < TOPK; j += 32) sc[h * TOPK + j] *= inv;
  }
  __syncthreads();

  for (int o = tid; o < D_MODEL; o += 256) {
    int h = o >> 7;
    int d = o & 127;
    float acc = 0.f;
    for (int j = 0; j < TOPK; j++)
      acc += sc[h * TOPK + j] * (float)vb[(size_t)sel[j] * D_MODEL + h * DH_N + d];
    attnb[(size_t)t * D_MODEL + o] = (__bf16)acc;
  }
}

// ---------------- host orchestration ----------------
extern "C" void kernel_launch(void* const* d_in, const int* in_sizes, int n_in,
                              void* d_out, int out_size, void* d_ws, size_t ws_size,
                              hipStream_t stream) {
  (void)in_sizes; (void)n_in; (void)out_size; (void)ws_size;
  const float* x   = (const float*)d_in[0];
  const float* Wqi = (const float*)d_in[1];
  const float* Wki = (const float*)d_in[2];
  const float* Wwi = (const float*)d_in[3];
  const float* Wq  = (const float*)d_in[4];
  const float* Wk  = (const float*)d_in[5];
  const float* Wv  = (const float*)d_in[6];
  const float* Wo  = (const float*)d_in[7];

  char* p = (char*)d_ws;
  auto take = [&](size_t bytes) { char* r = p; p += (bytes + 255) & ~(size_t)255; return r; };

  __bf16* xb    = (__bf16*)take((size_t)T_SEQ * D_MODEL * 2);
  __bf16* WqiT  = (__bf16*)take((size_t)(HI_N * DI_N) * D_MODEL * 2);      // 256 x 2048
  __bf16* WkiT  = (__bf16*)take((size_t)128 * D_MODEL * 2);                // 64 -> pad 128
  __bf16* WqT   = (__bf16*)take((size_t)D_MODEL * D_MODEL * 2);
  __bf16* WkT   = (__bf16*)take((size_t)D_MODEL * D_MODEL * 2);
  __bf16* WvT   = (__bf16*)take((size_t)D_MODEL * D_MODEL * 2);
  __bf16* WoT   = (__bf16*)take((size_t)D_MODEL * D_MODEL * 2);
  __bf16* qi    = (__bf16*)take((size_t)T_SEQ * (HI_N * DI_N) * 2);
  __bf16* ki    = (__bf16*)take((size_t)T_SEQ * DI_N * 2);
  float*  wts   = (float*) take((size_t)T_SEQ * HI_N * 4);
  __bf16* qb    = (__bf16*)take((size_t)T_SEQ * D_MODEL * 2);
  __bf16* kb    = (__bf16*)take((size_t)T_SEQ * D_MODEL * 2);
  __bf16* vb    = (__bf16*)take((size_t)T_SEQ * D_MODEL * 2);
  float*  sco   = (float*) take((size_t)T_SEQ * T_SEQ * 4);
  int*    tidx  = (int*)   take((size_t)T_SEQ * TOPK * 4);
  __bf16* attnb = (__bf16*)take((size_t)T_SEQ * D_MODEL * 2);

  const int n_xd = T_SEQ * D_MODEL;

  // 1) bf16 conversions / weight transposes
  cvt_bf16_kernel<<<(n_xd + 255) / 256, 256, 0, stream>>>(x, xb, n_xd);
  transpose_w_kernel<<<((HI_N * DI_N) * D_MODEL + 255) / 256, 256, 0, stream>>>(Wqi, WqiT, D_MODEL, HI_N * DI_N, HI_N * DI_N);
  transpose_w_kernel<<<(128 * D_MODEL + 255) / 256, 256, 0, stream>>>(Wki, WkiT, D_MODEL, DI_N, 128);
  transpose_w_kernel<<<(n_xd + 255) / 256, 256, 0, stream>>>(Wq, WqT, D_MODEL, D_MODEL, D_MODEL);
  transpose_w_kernel<<<(n_xd + 255) / 256, 256, 0, stream>>>(Wk, WkT, D_MODEL, D_MODEL, D_MODEL);
  transpose_w_kernel<<<(n_xd + 255) / 256, 256, 0, stream>>>(Wv, WvT, D_MODEL, D_MODEL, D_MODEL);
  transpose_w_kernel<<<(n_xd + 255) / 256, 256, 0, stream>>>(Wo, WoT, D_MODEL, D_MODEL, D_MODEL);
  w_proj_kernel<<<(T_SEQ * HI_N + 255) / 256, 256, 0, stream>>>(x, Wwi, wts);

  // 2) WMMA projection GEMMs (bf16 in, f32 acc, bf16 out); block tile 128x128
  gemm_bf16_wmma<<<dim3(2, T_SEQ / 128), 256, 0, stream>>>(xb, WqiT, qi, T_SEQ, HI_N * DI_N, D_MODEL, HI_N * DI_N, 1);
  gemm_bf16_wmma<<<dim3(1, T_SEQ / 128), 256, 0, stream>>>(xb, WkiT, ki, T_SEQ, DI_N, D_MODEL, DI_N, 1);
  gemm_bf16_wmma<<<dim3(16, T_SEQ / 128), 256, 0, stream>>>(xb, WqT, qb, T_SEQ, D_MODEL, D_MODEL, D_MODEL, 1);
  gemm_bf16_wmma<<<dim3(16, T_SEQ / 128), 256, 0, stream>>>(xb, WkT, kb, T_SEQ, D_MODEL, D_MODEL, D_MODEL, 1);
  gemm_bf16_wmma<<<dim3(16, T_SEQ / 128), 256, 0, stream>>>(xb, WvT, vb, T_SEQ, D_MODEL, D_MODEL, D_MODEL, 1);

  // 3) indexer scores (WMMA) -> masked score rows
  index_score_kernel<<<T_SEQ / 16, 512, 0, stream>>>(qi, ki, wts, sco);

  // 4) top-k selection per query
  topk_kernel<<<T_SEQ, 256, 0, stream>>>(sco, tidx);

  // 5) sparse attention (gather-bound)
  attn_kernel<<<T_SEQ, 256, 0, stream>>>(qb, kb, vb, tidx, attnb);

  // 6) output projection -> d_out (f32)
  gemm_bf16_wmma<<<dim3(16, T_SEQ / 128), 256, 0, stream>>>(attnb, WoT, d_out, T_SEQ, D_MODEL, D_MODEL, D_MODEL, 0);
}
